// SSGCL_90838558311071
// MI455X (gfx1250) — compile-verified, matrix operand order checked by reading, not run on hardware
//
#include <hip/hip_runtime.h>
#include <math.h>

// ---------------- static problem dims (match reference) ----------------
#define B_G   256
#define NPG   128
#define NT    (B_G*NPG)        // 32768 nodes
#define F_INN 128
#define DEG   16
#define E0C   (B_G*NPG*DEG)    // 524288 edges
#define ETC   (E0C + NT)       // + self loops = 557056
#define HH    4                // GAT heads
#define KD    20               // KAF dict
#define KPOOL 30
#define C1L   256
#define C2L   128
#define C3L   64
#define TOTC  448
#define DENSE_IN 352
#define KAF_GAMMA 11.28125f    // 1/(2*(4/19)^2)

typedef __attribute__((ext_vector_type(16))) _Float16 v16h;
typedef __attribute__((ext_vector_type(8)))  float    v8f;
typedef __attribute__((ext_vector_type(4)))  unsigned uv4;

__device__ __forceinline__ unsigned pk2(float a, float b) {
    union { _Float16 h[2]; unsigned u; } t;
    t.h[0] = (_Float16)a; t.h[1] = (_Float16)b;
    return t.u;
}
// relative even k (0..31) -> (half, pair) in WMMA 16-bit fragment layout
__device__ __forceinline__ void kmap(int k, int& hf, int& p) {
    if (k < 16) { hf = k >> 3; p = (k & 7) >> 1; }
    else        { int k2 = k - 16; hf = k2 >> 3; p = 4 + ((k2 & 7) >> 1); }
}

// Stage one 64x32 A tile and one 32x32 B tile into fragment-layout LDS.
// GK: guard k against K (remainder tile). GN: guard column against Nc.
template<int TB, bool GK, bool GN>
__device__ __forceinline__ void stage_tiles(
    const float* __restrict__ A, const float* __restrict__ Bm,
    int K, int Nc, int lda, int ldb, int kt, int row0, int col0, int tid,
    unsigned sAf[4][32][8], unsigned sBf[2][32][8])
{
    float2 av[8];
    #pragma unroll
    for (int i = 0; i < 8; ++i) {
        int pp = i*128 + tid;
        int r = pp >> 4, pc = pp & 15;
        int k = kt + 2*pc;
        if (!GK || k < K) av[i] = *(const float2*)&A[(long long)(row0 + r)*lda + k];
        else              av[i] = make_float2(0.f, 0.f);
    }
    float2 bv[4];
    #pragma unroll
    for (int i = 0; i < 4; ++i) {
        int pp = i*128 + tid;
        int n = pp & 31, kp = pp >> 5;
        int k = kt + 2*kp;
        int col = col0 + n;
        bool ok = (!GK || k < K) && (!GN || col < Nc);
        float2 v = make_float2(0.f, 0.f);
        if (ok) {
            if (TB) {
                v = *(const float2*)&Bm[(long long)col*ldb + k];
            } else {
                v.x = Bm[(long long)k*ldb + col];
                v.y = Bm[(long long)(k+1)*ldb + col];
            }
        }
        bv[i] = v;
    }
    #pragma unroll
    for (int i = 0; i < 8; ++i) {
        int pp = i*128 + tid;
        int r = pp >> 4, pc = pp & 15;
        int hf, p; kmap(2*pc, hf, p);
        sAf[r >> 4][hf*16 + (r & 15)][p] = pk2(av[i].x, av[i].y);
    }
    #pragma unroll
    for (int i = 0; i < 4; ++i) {
        int pp = i*128 + tid;
        int n = pp & 31, kp = pp >> 5;
        int hf, p; kmap(2*kp, hf, p);
        sBf[n >> 4][hf*16 + (n & 15)][p] = pk2(bv[i].x, bv[i].y);
    }
}

// ======================= generic WMMA f16 GEMM =========================
// C[M,Nc] = act( A[M,K] @ B(K,Nc) + bias ); TB: B stored [Nc,K].
// Batched via gridDim.z. Requires M%64==0, K even, lda/ldb even.
// Block: 128 thr (4 waves) computes a 64 x 32 output tile.
template<int TB>
__global__ void __launch_bounds__(128)
gemm_wmma_f16(const float* __restrict__ A, const float* __restrict__ Bm,
              const float* __restrict__ bias, float* __restrict__ C,
              int M, int K, int Nc, int lda, int ldb, int ldc, int act,
              long long strideA, long long strideB, long long strideC)
{
    __shared__ __align__(16) unsigned sAf[4][32][8];   // per-wave A fragment
    __shared__ __align__(16) unsigned sBf[2][32][8];   // two 16-col B fragments

    const long long bz = blockIdx.z;
    A  += bz * strideA;
    Bm += bz * strideB;
    C  += bz * strideC;

    const int tid  = threadIdx.x;
    const int wave = tid >> 5;
    const int lane = tid & 31;
    const int mrow = lane & 15;
    const int half = lane >> 4;
    const int row0 = blockIdx.y * 64;
    const int col0 = blockIdx.x * 32;
    const bool fullN = (Nc - col0) >= 32;
    const bool has2  = (Nc - col0) > 16;

    v8f acc0 = {0.f,0.f,0.f,0.f,0.f,0.f,0.f,0.f};
    v8f acc1 = {0.f,0.f,0.f,0.f,0.f,0.f,0.f,0.f};

    const uv4* pa  = (const uv4*)&sAf[wave][lane][0];
    const uv4* pb0 = (const uv4*)&sBf[0][lane][0];
    const uv4* pb1 = (const uv4*)&sBf[1][lane][0];

    int kt = 0;
    #define GEMM_STEP()                                                        \
        do {                                                                   \
            __syncthreads();                                                   \
            union FR { uv4 q[2]; v16h v; } fa, fb0, fb1;                       \
            fa.q[0]  = pa[0];  fa.q[1]  = pa[1];                               \
            fb0.q[0] = pb0[0]; fb0.q[1] = pb0[1];                              \
            fb1.q[0] = pb1[0]; fb1.q[1] = pb1[1];                              \
            acc0 = __builtin_amdgcn_wmma_f32_16x16x32_f16(                     \
                       false, fa.v, false, fb0.v, (short)0, acc0, false, false);\
            acc1 = __builtin_amdgcn_wmma_f32_16x16x32_f16(                     \
                       false, fa.v, false, fb1.v, (short)0, acc1, false, false);\
            __syncthreads();                                                   \
        } while (0)

    if (fullN) {
        for (; kt + 32 <= K; kt += 32) {
            stage_tiles<TB, false, false>(A, Bm, K, Nc, lda, ldb, kt, row0, col0, tid, sAf, sBf);
            GEMM_STEP();
        }
    } else {
        for (; kt + 32 <= K; kt += 32) {
            stage_tiles<TB, false, true>(A, Bm, K, Nc, lda, ldb, kt, row0, col0, tid, sAf, sBf);
            GEMM_STEP();
        }
    }
    if (kt < K) {   // remainder K tile (only K%32 != 0 call sites)
        stage_tiles<TB, true, true>(A, Bm, K, Nc, lda, ldb, kt, row0, col0, tid, sAf, sBf);
        GEMM_STEP();
    }
    #undef GEMM_STEP

    // ---- epilogue ----
    #pragma unroll
    for (int vg = 0; vg < 8; ++vg) {
        int r  = row0 + wave*16 + half*8 + vg;
        int cc = col0 + mrow;
        float v = acc0[vg];
        if (bias) v += bias[cc];
        if (act == 1) v = fmaxf(v, 0.0f);
        else if (act == 2) v = 1.0f / (1.0f + __expf(-v));
        C[(long long)r*ldc + cc] = v;
    }
    if (has2) {
        #pragma unroll
        for (int vg = 0; vg < 8; ++vg) {
            int r  = row0 + wave*16 + half*8 + vg;
            int cc = col0 + 16 + mrow;
            float v = acc1[vg];
            if (bias) v += bias[cc];
            if (act == 1) v = fmaxf(v, 0.0f);
            else if (act == 2) v = 1.0f / (1.0f + __expf(-v));
            C[(long long)r*ldc + cc] = v;
        }
    }
}

// ======================= elementwise / utility =========================
__global__ void fill_f32(float* p, float v, long long n) {
    long long i = (long long)blockIdx.x*blockDim.x + threadIdx.x;
    if (i < n) p[i] = v;
}
__global__ void fill_u32(unsigned* p, unsigned v, long long n) {
    long long i = (long long)blockIdx.x*blockDim.x + threadIdx.x;
    if (i < n) p[i] = v;
}
__global__ void init_bias_k(float* out, const float* __restrict__ b, long long total, int C) {
    long long i = (long long)blockIdx.x*blockDim.x + threadIdx.x;
    if (i < total) out[i] = b[i % C];
}
__global__ void sigmoid_k(float* p, long long n) {
    long long i = (long long)blockIdx.x*blockDim.x + threadIdx.x;
    if (i < n) p[i] = 1.0f / (1.0f + __expf(-p[i]));
}
__global__ void mul_k(float* a, const float* __restrict__ b, long long n) {
    long long i = (long long)blockIdx.x*blockDim.x + threadIdx.x;
    if (i < n) a[i] *= b[i];
}
__global__ void combine_k(const float* __restrict__ f, const float* __restrict__ g,
                          const float* __restrict__ lin, float* out, long long n) {
    long long i = (long long)blockIdx.x*blockDim.x + threadIdx.x;
    if (i < n) out[i] = f[i]*g[i] + lin[i];
}
__global__ void kaf_k(const float* __restrict__ in, const float* __restrict__ alpha,
                      float* out, long long total, int C) {
    long long i = (long long)blockIdx.x*blockDim.x + threadIdx.x;
    if (i >= total) return;
    int cc = (int)(i % C);
    float x = in[i];
    const float* al = alpha + (long long)cc*KD;
    float s = 0.0f;
    #pragma unroll
    for (int d = 0; d < KD; ++d) {
        float kd = -2.0f + d*(4.0f/19.0f);
        float t = x - kd;
        s += al[d] * __expf(-KAF_GAMMA * t * t);
    }
    out[i] = s;
}

// ======================= GAT pieces =========================
__global__ void gat_logits_k(const float* __restrict__ h, const float* __restrict__ as,
                             const float* __restrict__ ad, float* es, float* ed, int C) {
    int id = blockIdx.x*blockDim.x + threadIdx.x;
    if (id >= NT*HH) return;
    int n = id / HH, hh = id % HH;
    const float* hp = h + (long long)n*HH*C + (long long)hh*C;
    const float* ap = as + (long long)hh*C;
    const float* bp = ad + (long long)hh*C;
    float s = 0.f, d = 0.f;
    for (int j = 0; j < C; ++j) { float v = hp[j]; s += v*ap[j]; d += v*bp[j]; }
    es[id] = s; ed[id] = d;
}
__device__ __forceinline__ unsigned fkey(float f) {
    unsigned u = __float_as_uint(f);
    return (u & 0x80000000u) ? ~u : (u | 0x80000000u);
}
__device__ __forceinline__ float funkey(unsigned k) {
    return __uint_as_float((k & 0x80000000u) ? (k ^ 0x80000000u) : ~k);
}
__global__ void edge_max_k(const int* __restrict__ src, const int* __restrict__ dst,
                           const float* __restrict__ es, const float* __restrict__ ed,
                           unsigned* mkey) {
    long long id = (long long)blockIdx.x*blockDim.x + threadIdx.x;
    if (id >= (long long)ETC*HH) return;
    int e = (int)(id / HH), hh = (int)(id % HH);
    int s = (e < E0C) ? src[e] : (e - E0C);
    int d = (e < E0C) ? dst[e] : (e - E0C);
    float v = es[s*HH+hh] + ed[d*HH+hh];
    v = (v > 0.f) ? v : 0.2f*v;                 // leaky_relu(0.2)
    atomicMax(&mkey[d*HH+hh], fkey(v));
}
__global__ void edge_expsum_k(const int* __restrict__ src, const int* __restrict__ dst,
                              const float* __restrict__ es, const float* __restrict__ ed,
                              const unsigned* __restrict__ mkey, float* ssum, float* wedge) {
    long long id = (long long)blockIdx.x*blockDim.x + threadIdx.x;
    if (id >= (long long)ETC*HH) return;
    int e = (int)(id / HH), hh = (int)(id % HH);
    int s = (e < E0C) ? src[e] : (e - E0C);
    int d = (e < E0C) ? dst[e] : (e - E0C);
    float v = es[s*HH+hh] + ed[d*HH+hh];
    v = (v > 0.f) ? v : 0.2f*v;
    float w = __expf(v - funkey(mkey[d*HH+hh]));
    wedge[id] = w;
    atomicAdd(&ssum[d*HH+hh], w);
}
// one block per edge, blockDim = C; accumulate head-averaged message into out[dst]
__global__ void gat_agg_k(const int* __restrict__ src, const int* __restrict__ dst,
                          const float* __restrict__ h, const float* __restrict__ wedge,
                          const float* __restrict__ ssum, float* out, int C) {
    int e = blockIdx.x;
    int cc = threadIdx.x;
    int s = (e < E0C) ? src[e] : (e - E0C);
    int d = (e < E0C) ? dst[e] : (e - E0C);
    float a = 0.f;
    #pragma unroll
    for (int hh = 0; hh < HH; ++hh) {
        float alpha = wedge[(long long)e*HH+hh] / (ssum[d*HH+hh] + 1e-16f);
        a += alpha * h[(long long)s*HH*C + (long long)hh*C + cc];
    }
    atomicAdd(&out[(long long)d*C + cc], a * (1.0f/HH));
}

// ======================= BatchNorm (batch stats, in place) =============
__global__ void bn_k(float* x, const float* __restrict__ g, const float* __restrict__ b, int C) {
    int cc = blockIdx.x;
    int t = threadIdx.x;
    __shared__ float ss[256], sq[256];
    __shared__ float mu, inv;
    float s = 0.f, q = 0.f;
    for (int n = t; n < NT; n += 256) {
        float v = x[(long long)n*C + cc]; s += v; q += v*v;
    }
    ss[t] = s; sq[t] = q; __syncthreads();
    for (int st = 128; st > 0; st >>= 1) {
        if (t < st) { ss[t] += ss[t+st]; sq[t] += sq[t+st]; }
        __syncthreads();
    }
    if (t == 0) {
        float m = ss[0] / (float)NT;
        float var = sq[0] / (float)NT - m*m;
        mu = m; inv = rsqrtf(var + 1e-5f);
    }
    __syncthreads();
    float gg = g[cc], bb = b[cc];
    for (int n = t; n < NT; n += 256) {
        long long idx = (long long)n*C + cc;
        x[idx] = (x[idx] - mu) * inv * gg + bb;
    }
}

// ======================= attention softmax (rows of 128) ===============
__global__ void softmax128_k(float* sc, float scale) {
    int row = blockIdx.x;
    int t = threadIdx.x;
    float* p = sc + (long long)row*NPG;
    __shared__ float sh[NPG];
    float v = p[t] * scale;
    sh[t] = v; __syncthreads();
    for (int st = 64; st > 0; st >>= 1) { if (t < st) sh[t] = fmaxf(sh[t], sh[t+st]); __syncthreads(); }
    float mx = sh[0]; __syncthreads();
    float e = __expf(v - mx);
    sh[t] = e; __syncthreads();
    for (int st = 64; st > 0; st >>= 1) { if (t < st) sh[t] += sh[t+st]; __syncthreads(); }
    float s = sh[0];
    p[t] = e / s;
}

// ======================= tail: concat / sortpool / convs ===============
__global__ void concat_k(const float* __restrict__ x1, const float* __restrict__ x2,
                         const float* __restrict__ x3, float* cs) {
    long long id = (long long)blockIdx.x*blockDim.x + threadIdx.x;
    if (id >= (long long)NT*TOTC) return;
    int j = (int)(id % TOTC);
    long long n = id / TOTC;
    float v;
    if (j < C1L)            v = x1[n*C1L + j];
    else if (j < C1L+C2L)   v = x2[n*C2L + (j - C1L)];
    else                    v = x3[n*C3L + (j - C1L - C2L)];
    cs[id] = v;
}
__global__ void sortpool_k(const float* __restrict__ cs, int* idx) {
    int g = blockIdx.x, t = threadIdx.x;     // 128 threads
    __shared__ float vals[NPG];
    vals[t] = cs[((long long)g*NPG + t)*TOTC + (TOTC-1)];
    __syncthreads();
    float v = vals[t];
    int rank = 0;
    for (int j = 0; j < NPG; ++j) {
        float vj = vals[j];
        rank += (vj > v) || (vj == v && j < t);   // stable descending
    }
    if (rank < KPOOL) idx[g*KPOOL + rank] = t;
}
__global__ void gather_k(const float* __restrict__ cs, const int* __restrict__ idx, float* pl) {
    long long id = (long long)blockIdx.x*blockDim.x + threadIdx.x;
    if (id >= (long long)B_G*KPOOL*TOTC) return;
    int j = (int)(id % TOTC);
    long long bk = id / TOTC;
    int b = (int)(bk / KPOOL), kk = (int)(bk % KPOOL);
    int node = idx[b*KPOOL + kk];
    pl[id] = cs[((long long)b*NPG + node)*TOTC + j];
}
__global__ void maxpool_k(const float* __restrict__ c1o, float* p1) {
    int id = blockIdx.x*blockDim.x + threadIdx.x;   // B*16*15
    if (id >= B_G*16*15) return;
    int t = id % 15; int tmp = id / 15;
    int cch = tmp % 16; int b = tmp / 16;
    float a  = c1o[((long long)b*KPOOL + 2*t  )*16 + cch];
    float b2 = c1o[((long long)b*KPOOL + 2*t+1)*16 + cch];
    p1[((long long)b*16 + cch)*15 + t] = fmaxf(a, b2);
}
__global__ void conv2_k(const float* __restrict__ p1, const float* __restrict__ w,
                        const float* __restrict__ bias, float* c2o) {
    int id = blockIdx.x*blockDim.x + threadIdx.x;   // B*32*11
    if (id >= B_G*32*11) return;
    int t = id % 11; int tmp = id / 11;
    int o = tmp % 32; int b = tmp / 32;
    float s = bias[o];
    for (int i = 0; i < 16; ++i)
        #pragma unroll
        for (int kk = 0; kk < 5; ++kk)
            s += p1[((long long)b*16 + i)*15 + t + kk] * w[(o*16 + i)*5 + kk];
    c2o[((long long)b*32 + o)*11 + t] = fmaxf(s, 0.0f);
}
__global__ void lin2_k(const float* __restrict__ fl1, const float* __restrict__ w,
                       const float* __restrict__ b, float* out) {
    int bb = blockIdx.x*blockDim.x + threadIdx.x;
    if (bb >= B_G) return;
    float s = b[0];
    for (int j = 0; j < 128; ++j) s += fl1[bb*128 + j] * w[j];
    out[bb] = s;
}

// ======================= host orchestration =========================
static inline long long cdivll(long long a, long long b) { return (a + b - 1) / b; }

extern "C" void kernel_launch(void* const* d_in, const int* in_sizes, int n_in,
                              void* d_out, int out_size, void* d_ws, size_t ws_size,
                              hipStream_t stream)
{
    (void)in_sizes; (void)n_in; (void)out_size; (void)ws_size;

    const float* X0  = (const float*)d_in[0];
    const int*   EIX = (const int*)d_in[1];
    const int*   src = EIX;
    const int*   dst = EIX + E0C;
    // d_in[2] = batch (unused: graphs are fixed-size contiguous)

    const float* LP[3][24];
    int pi = 3;
    for (int l = 0; l < 3; ++l)
        for (int j = 0; j < 24; ++j) LP[l][j] = (const float*)d_in[pi++];
    const float* c1_w   = (const float*)d_in[pi++];
    const float* c1_b   = (const float*)d_in[pi++];
    const float* c2_w   = (const float*)d_in[pi++];
    const float* c2_b   = (const float*)d_in[pi++];
    const float* lin1_w = (const float*)d_in[pi++];
    const float* lin1_b = (const float*)d_in[pi++];
    const float* lin2_w = (const float*)d_in[pi++];
    const float* lin2_b = (const float*)d_in[pi++];

    float* ws = (float*)d_ws;
    // ---- workspace map (float offsets) ----
    const long long SZ_HBIG = (long long)NT*HH*C1L;                 // 33.5M
    const long long NB      = (long long)NT*C1L;                    // 8.39M
    float*    HB   = ws;
    float*    B0   = ws + SZ_HBIG;
    float*    B1   = B0 + NB;
    float*    B2   = B1 + NB;
    float*    B3   = B2 + NB;
    float*    T1   = B3 + NB;                                       // NT*64
    float*    ES   = T1 + (long long)NT*64;
    float*    ED   = ES + (long long)NT*HH;
    unsigned* MK   = (unsigned*)(ED + (long long)NT*HH);
    float*    SS   = (float*)(MK + (long long)NT*HH);
    float*    WE   = SS + (long long)NT*HH;                         // ET*H
    float*    SC   = WE + (long long)ETC*HH;                        // B*128*128
    float*    X1   = SC + (long long)B_G*NPG*NPG;
    float*    X2   = X1 + (long long)NT*C1L;
    float*    X3   = X2 + (long long)NT*C2L;
    // tail aliases into HB region (GAT projections dead by then)
    float* CS  = HB;                                                // NT*448
    float* PL  = CS  + (long long)NT*TOTC;                          // B*30*448
    float* C1O = PL  + (long long)B_G*KPOOL*TOTC;                   // B*30*16
    float* P1  = C1O + (long long)B_G*KPOOL*16;                     // B*240
    float* C2O = P1  + (long long)B_G*16*15;                        // B*352
    float* FL1 = C2O + (long long)B_G*32*11;                        // B*128
    int*   IDX = (int*)(FL1 + (long long)B_G*128);                  // B*30

    const dim3 blk128(128);
    auto gemm = [&](const float* A, const float* Bm, const float* bias, float* C,
                    int M, int K, int Nc, int lda, int ldb, int ldc,
                    int tB, int act, long long sA, long long sB, long long sCc, int batches) {
        dim3 grid((unsigned)cdivll(Nc, 32), M/64, batches);
        if (tB)
            gemm_wmma_f16<1><<<grid, blk128, 0, stream>>>(A, Bm, bias, C, M, K, Nc,
                                                          lda, ldb, ldc, act, sA, sB, sCc);
        else
            gemm_wmma_f16<0><<<grid, blk128, 0, stream>>>(A, Bm, bias, C, M, K, Nc,
                                                          lda, ldb, ldc, act, sA, sB, sCc);
    };
    auto ew = [&](long long n) { return dim3((unsigned)cdivll(n, 256)); };

    const float* Xin = X0;
    float* Xouts[3] = {X1, X2, X3};
    const int fins[3] = {F_INN, C1L, C2L};
    const int cs_[3]  = {C1L, C2L, C3L};

    for (int l = 0; l < 3; ++l) {
        const int fin = fins[l], c = cs_[l], r = c/4;
        const float* const* P = LP[l];
        const long long nc = (long long)NT*c;
        float* Xout = Xouts[l];

        // ---------- GAT feature branch ----------
        gemm(Xin, P[0], nullptr, HB, NT, fin, HH*c, fin, HH*c, HH*c, 0, 0, 0,0,0, 1);
        gat_logits_k<<<ew((long long)NT*HH), 256, 0, stream>>>(HB, P[1], P[2], ES, ED, c);
        fill_u32<<<ew((long long)NT*HH), 256, 0, stream>>>(MK, 0u, (long long)NT*HH);
        fill_f32<<<ew((long long)NT*HH), 256, 0, stream>>>(SS, 0.f, (long long)NT*HH);
        edge_max_k<<<ew((long long)ETC*HH), 256, 0, stream>>>(src, dst, ES, ED, MK);
        edge_expsum_k<<<ew((long long)ETC*HH), 256, 0, stream>>>(src, dst, ES, ED, MK, SS, WE);
        init_bias_k<<<ew(nc), 256, 0, stream>>>(B0, P[3], nc, c);
        gat_agg_k<<<dim3(ETC), dim3(c), 0, stream>>>(src, dst, HB, WE, SS, B0, c);
        kaf_k<<<ew(nc), 256, 0, stream>>>(B0, P[12], B0, nc, c);       // kaf_in

        // ---------- GAT gate branch ----------
        gemm(Xin, P[4], nullptr, HB, NT, fin, HH*c, fin, HH*c, HH*c, 0, 0, 0,0,0, 1);
        gat_logits_k<<<ew((long long)NT*HH), 256, 0, stream>>>(HB, P[5], P[6], ES, ED, c);
        fill_u32<<<ew((long long)NT*HH), 256, 0, stream>>>(MK, 0u, (long long)NT*HH);
        fill_f32<<<ew((long long)NT*HH), 256, 0, stream>>>(SS, 0.f, (long long)NT*HH);
        edge_max_k<<<ew((long long)ETC*HH), 256, 0, stream>>>(src, dst, ES, ED, MK);
        edge_expsum_k<<<ew((long long)ETC*HH), 256, 0, stream>>>(src, dst, ES, ED, MK, SS, WE);
        init_bias_k<<<ew(nc), 256, 0, stream>>>(B1, P[7], nc, c);
        gat_agg_k<<<dim3(ETC), dim3(c), 0, stream>>>(src, dst, HB, WE, SS, B1, c);
        sigmoid_k<<<ew(nc), 256, 0, stream>>>(B1, nc);

        // ---------- linear skip + combine + BN ----------
        gemm(Xin, P[8], P[9], B2, NT, fin, c, fin, c, c, 0, 0, 0,0,0, 1);
        combine_k<<<ew(nc), 256, 0, stream>>>(B0, B1, B2, B3, nc);
        bn_k<<<dim3(c), dim3(256), 0, stream>>>(B3, P[10], P[11], c);

        // ---------- kaf_out + semantic gating ----------
        kaf_k<<<ew(nc), 256, 0, stream>>>(B3, P[13], B0, nc, c);       // kaf_out
        gemm(B0, P[14], P[15], T1, NT, c, r, c, r, r, 0, 1, 0,0,0, 1); // relu
        gemm(T1, P[16], P[17], B1, NT, r, c, r, c, c, 0, 2, 0,0,0, 1); // sigmoid
        mul_k<<<ew(nc), 256, 0, stream>>>(B0, B1, nc);

        // ---------- structural (per-graph) attention ----------
        gemm(B0, P[18], P[19], B1, NT, c, c, c, c, c, 0, 0, 0,0,0, 1); // q
        gemm(B0, P[20], P[21], B2, NT, c, c, c, c, c, 0, 0, 0,0,0, 1); // k
        gemm(B0, P[22], P[23], B3, NT, c, c, c, c, c, 0, 0, 0,0,0, 1); // v
        gemm(B1, B2, nullptr, SC, NPG, c, NPG, c, c, NPG, 1, 0,
             (long long)NPG*c, (long long)NPG*c, (long long)NPG*NPG, B_G);   // q @ k^T
        softmax128_k<<<dim3(B_G*NPG), dim3(NPG), 0, stream>>>(SC, 1.0f/sqrtf((float)c));
        gemm(SC, B3, nullptr, Xout, NPG, NPG, c, NPG, c, c, 0, 0,
             (long long)NPG*NPG, (long long)NPG*c, (long long)NPG*c, B_G);   // a @ v

        Xin = Xout;
    }

    // ---------------- tail ----------------
    concat_k<<<ew((long long)NT*TOTC), 256, 0, stream>>>(X1, X2, X3, CS);
    sortpool_k<<<dim3(B_G), dim3(NPG), 0, stream>>>(CS, IDX);
    gather_k<<<ew((long long)B_G*KPOOL*TOTC), 256, 0, stream>>>(CS, IDX, PL);
    // conv1 as GEMM: [B*30,448] @ c1_w^T[448,16], relu
    gemm(PL, c1_w, c1_b, C1O, B_G*KPOOL, TOTC, 16, TOTC, TOTC, 16, 1, 1, 0,0,0, 1);
    maxpool_k<<<ew((long long)B_G*16*15), 256, 0, stream>>>(C1O, P1);
    conv2_k<<<ew((long long)B_G*32*11), 256, 0, stream>>>(P1, c2_w, c2_b, C2O);
    // lin1: [256,352] @ [352,128], relu
    gemm(C2O, lin1_w, lin1_b, FL1, B_G, DENSE_IN, 128, DENSE_IN, 128, 128, 0, 1, 0,0,0, 1);
    lin2_k<<<ew(B_G), 256, 0, stream>>>(FL1, lin2_w, lin2_b, (float*)d_out);
}
